// FourierVSSBlock_38534446580134
// MI455X (gfx1250) — matrix operand order repaired and checked by reading, not compiled, not run on hardware
//
#include <hip/hip_runtime.h>
#include <hip/hip_bf16.h>
#include <hip/hip_fp16.h>

// ---------------------------------------------------------------------------
// FourierVSSBlock forward for MI455X (gfx1250).
// All GEMM-shaped math (1x1 conv, QKV proj, attention, out-proj) runs through
// V_WMMA_F32_16X16X32_F16 (f16 in, f32 accumulate). Everything fits in L2
// (192MB), so the rest (DFTs, 3x3 convs, KAN gates) is plain VALU code.
// ---------------------------------------------------------------------------

typedef __attribute__((ext_vector_type(16))) _Float16 v16h;
typedef __attribute__((ext_vector_type(8)))  float    v8f;

#define DEV static __device__ __forceinline__

#define BB   2
#define CC   32
#define C2C  64
#define HH_  64
#define WW_  64
#define NPIX 4096              // H*W
#define NROW (BB*NPIX)         // 8192 token rows
#define WF   33                // rfft width bins
#define PFREQ (HH_*WF)         // 2112 freq pixels
#define TWO_PI 6.28318530717958647692f

// ---- WMMA operand k maps (contiguous k = e + 16*hi, same for A and B) --------
DEV int a_kmap(int e, int hi) { return e + (hi ? 16 : 0); }
DEV int b_kmap(int e, int hi) { return e + (hi ? 16 : 0); }

DEV v8f wmma_f16(v16h a, v16h b, v8f c) {
  return __builtin_amdgcn_wmma_f32_16x16x32_f16(false, a, false, b, (short)0, c,
                                                false, false);
}

DEV float sigmoidf_(float x) { return 1.f / (1.f + __expf(-x)); }
DEV float siluf_(float x)    { return x * sigmoidf_(x); }
DEV float geluf_(float x)    { return 0.5f * x * (1.f + erff(x * 0.70710678118654752f)); }
DEV float leakyf_(float x)   { return (x >= 0.f) ? x : 0.01f * x; }

// Cox-de Boor B-spline bases (SO=3, GS=10): 13 outputs.
DEV void bspline13(float x, float* out) {
  float g[17];
#pragma unroll
  for (int i = 0; i < 17; ++i) g[i] = (float)(i - 3) * 0.2f - 1.0f;
  float b[16];
#pragma unroll
  for (int i = 0; i < 16; ++i) b[i] = (x >= g[i] && x < g[i + 1]) ? 1.f : 0.f;
#pragma unroll
  for (int p = 1; p <= 3; ++p) {
    for (int i = 0; i < 16 - p; ++i) {
      float l = (x - g[i]) / (g[i + p] - g[i]) * b[i];
      float r = (g[i + p + 1] - x) / (g[i + p + 1] - g[i + 1]) * b[i + 1];
      b[i] = l + r;
    }
  }
#pragma unroll
  for (int i = 0; i < 13; ++i) out[i] = b[i];
}

// ---------------------------------------------------------------------------
// 1) stem: out[b,p,o] = sum_{i<64} w[o,i] * cat(input_x,input_b)[b,i,p]
//    WMMA: M=pixel rows, K=64 (two 32-steps), N=32. Output BHWC f32.
// ---------------------------------------------------------------------------
__global__ void k_conv1x1(const float* __restrict__ xa, const float* __restrict__ xb,
                          const float* __restrict__ w, float* __restrict__ out) {
  int lane = threadIdx.x & 31, hi = lane >> 4, lr = lane & 15;
  int row0 = blockIdx.x * 16;
  v8f c0 = {}, c1 = {};
#pragma unroll
  for (int kt = 0; kt < 2; ++kt) {
    const float* src = kt ? xb : xa;
    v16h a;
#pragma unroll
    for (int e = 0; e < 16; ++e) {
      int k = a_kmap(e, hi);
      int row = row0 + lr;
      int b = row >> 12, p = row & 4095;
      a[e] = (_Float16)src[((size_t)b * CC + k) * NPIX + p];
    }
    v16h bm;
#pragma unroll
    for (int e = 0; e < 16; ++e)
      bm[e] = (_Float16)w[(0 * 16 + lr) * C2C + kt * 32 + b_kmap(e, hi)];
    c0 = wmma_f16(a, bm, c0);
#pragma unroll
    for (int e = 0; e < 16; ++e)
      bm[e] = (_Float16)w[(16 + lr) * C2C + kt * 32 + b_kmap(e, hi)];
    c1 = wmma_f16(a, bm, c1);
  }
#pragma unroll
  for (int r = 0; r < 8; ++r) {
    int rg = row0 + r + 8 * hi;
    out[(size_t)rg * CC + lr] = c0[r];
    out[(size_t)rg * CC + 16 + lr] = c1[r];
  }
}

// ---------------------------------------------------------------------------
// 2) LayerNorm over 32 channels (one wave per pixel, lane = channel).
// ---------------------------------------------------------------------------
__global__ void k_layernorm32(const float* __restrict__ x, const float* __restrict__ g,
                              const float* __restrict__ bt, float* __restrict__ o) {
  int lane = threadIdx.x & 31;
  int pix = blockIdx.x * (blockDim.x >> 5) + (threadIdx.x >> 5);
  float v = x[(size_t)pix * CC + lane];
  float s = v;
#pragma unroll
  for (int off = 16; off; off >>= 1) s += __shfl_xor(s, off);
  float m = s * (1.f / 32.f);
  float d = v - m;
  float vs = d * d;
#pragma unroll
  for (int off = 16; off; off >>= 1) vs += __shfl_xor(vs, off);
  float inv = rsqrtf(vs * (1.f / 32.f) + 1e-6f);
  o[(size_t)pix * CC + lane] = d * inv * g[lane] + bt[lane];
}

// ---------------------------------------------------------------------------
// 3) Global average pool over pixels. bhwc: x[(b*P+p)*C+c] else x[(b*C+c)*P+p]
// ---------------------------------------------------------------------------
__global__ void k_pool(const float* __restrict__ x, float* __restrict__ pooled,
                       int C, int P, int bhwc) {
  int bc = blockIdx.x, b = bc / C, c = bc - b * C;
  float s = 0.f;
  for (int p = threadIdx.x; p < P; p += blockDim.x)
    s += bhwc ? x[((size_t)b * P + p) * C + c] : x[((size_t)b * C + c) * P + p];
  __shared__ float red[256];
  red[threadIdx.x] = s;
  __syncthreads();
  for (int o = 128; o; o >>= 1) {
    if (threadIdx.x < (unsigned)o) red[threadIdx.x] += red[threadIdx.x + o];
    __syncthreads();
  }
  if (threadIdx.x == 0) pooled[bc] = red[0] / (float)P;
}

// ---------------------------------------------------------------------------
// 4) KAN gate chain: gelu(fc) -> sigmoid(fc2) -> kan_linear -> kan_linear.
//    Tiny (B=2, C<=64): one block, threads = 2*C.
// ---------------------------------------------------------------------------
__global__ void k_kan_gates(const float* __restrict__ pooled,
                            const float* __restrict__ fw, const float* __restrict__ fb,
                            const float* __restrict__ f2w, const float* __restrict__ f2b,
                            const float* __restrict__ k1b, const float* __restrict__ k1s,
                            const float* __restrict__ k2b, const float* __restrict__ k2s,
                            float* __restrict__ outg, int C, int R) {
  int tid = threadIdx.x;
  int b = tid / C, o = tid - b * C;
  __shared__ float sm[BB][16];
  __shared__ float sact[BB][C2C];
  __shared__ float sbas[BB][C2C][13];
  __shared__ float sval[BB][C2C];

  if (o < R) {
    float a = fb[o];
    for (int c = 0; c < C; ++c) a += pooled[b * C + c] * fw[o * C + c];
    sm[b][o] = geluf_(a);
  }
  __syncthreads();
  {
    float a = f2b[o];
    for (int j = 0; j < R; ++j) a += sm[b][j] * f2w[o * R + j];
    sval[b][o] = sigmoidf_(a);
  }
  __syncthreads();
  {
    float t = sval[b][o];
    sact[b][o] = siluf_(t);
    bspline13(t, &sbas[b][o][0]);
  }
  __syncthreads();
  float u = 0.f;
  for (int i = 0; i < C; ++i) {
    u += sact[b][i] * k1b[o * C + i];
    const float* sp = &k1s[(o * C + i) * 13];
    for (int g = 0; g < 13; ++g) u += sbas[b][i][g] * sp[g];
  }
  __syncthreads();
  sval[b][o] = u;
  __syncthreads();
  {
    float t = sval[b][o];
    sact[b][o] = siluf_(t);
    bspline13(t, &sbas[b][o][0]);
  }
  __syncthreads();
  {
    float a = 0.f;
    for (int i = 0; i < C; ++i) {
      a += sact[b][i] * k2b[o * C + i];
      const float* sp = &k2s[(o * C + i) * 13];
      for (int g = 0; g < 13; ++g) a += sbas[b][i][g] * sp[g];
    }
    outg[b * C + o] = a;
  }
}

// out = x*(1+cs+ss) (+res). Same layout in/out, linear index.
__global__ void k_kan_apply(const float* __restrict__ x, const float* __restrict__ cs,
                            const float* __restrict__ ss, const float* __restrict__ res,
                            float* __restrict__ out, int C, int P, int bhwc) {
  int i = blockIdx.x * blockDim.x + threadIdx.x;
  int total = BB * C * P;
  if (i >= total) return;
  int c, b;
  if (bhwc) { c = i % C; int t = i / C; b = t / P; }
  else      { int t = i / P; c = t % C; b = t / C; }
  float g = 1.f + cs[b * C + c] + ss[b * C + c];
  float v = x[i] * g;
  out[i] = res ? (v + res[i]) : v;
}

// ---------------------------------------------------------------------------
// 5) 32x32 projection, f16 output:  O = A @ W  (A f32 [M,32], W f32 [32,32])
// ---------------------------------------------------------------------------
__global__ void k_proj32_f16(const float* __restrict__ A, const float* __restrict__ W,
                             _Float16* __restrict__ O) {
  int lane = threadIdx.x & 31, hi = lane >> 4, lr = lane & 15;
  int row0 = blockIdx.x * 16;
  v16h a;
#pragma unroll
  for (int e = 0; e < 16; ++e)
    a[e] = (_Float16)A[(size_t)(row0 + lr) * 32 + a_kmap(e, hi)];
#pragma unroll
  for (int t = 0; t < 2; ++t) {
    int n = t * 16 + lr;
    v16h bm;
#pragma unroll
    for (int e = 0; e < 16; ++e)
      bm[e] = (_Float16)W[b_kmap(e, hi) * 32 + n];
    v8f c = {};
    c = wmma_f16(a, bm, c);
#pragma unroll
    for (int r = 0; r < 8; ++r)
      O[(size_t)(row0 + r + 8 * hi) * 32 + n] = (_Float16)c[r];
  }
}

// O = A @ W + res (all f32 [M,32])
__global__ void k_outproj32(const float* __restrict__ A, const float* __restrict__ W,
                            const float* __restrict__ res, float* __restrict__ O) {
  int lane = threadIdx.x & 31, hi = lane >> 4, lr = lane & 15;
  int row0 = blockIdx.x * 16;
  v16h a;
#pragma unroll
  for (int e = 0; e < 16; ++e)
    a[e] = (_Float16)A[(size_t)(row0 + lr) * 32 + a_kmap(e, hi)];
#pragma unroll
  for (int t = 0; t < 2; ++t) {
    int n = t * 16 + lr;
    v16h bm;
#pragma unroll
    for (int e = 0; e < 16; ++e)
      bm[e] = (_Float16)W[b_kmap(e, hi) * 32 + n];
    v8f c = {};
    c = wmma_f16(a, bm, c);
#pragma unroll
    for (int r = 0; r < 8; ++r) {
      size_t idx = (size_t)(row0 + r + 8 * hi) * 32 + n;
      O[idx] = c[r] + res[idx];
    }
  }
}

// ---------------------------------------------------------------------------
// 6) Flash attention: per (b,h) wave handles 16 query rows, streams 32 keys
//    per iteration. d=8 padded into the WMMA K dimension. q/k/v f16 [B*N,32]
//    with column = head*8 + dd.
// ---------------------------------------------------------------------------
__global__ void k_attn_flash(const _Float16* __restrict__ Q, const _Float16* __restrict__ K,
                             const _Float16* __restrict__ V, float* __restrict__ O) {
  const int N = NPIX;
  int lane = threadIdx.x & 31, hi = lane >> 4, lr = lane & 15;
  int b = blockIdx.z, h = blockIdx.y, row0 = blockIdx.x * 16;
  const float scale = 0.35355339059327373f;  // 1/sqrt(8)
  size_t base = (size_t)b * N * 32;

  v16h aq;
#pragma unroll
  for (int e = 0; e < 16; ++e) {
    int k = a_kmap(e, hi);
    float qv = (k < 8) ? (float)Q[base + (size_t)(row0 + lr) * 32 + h * 8 + k] : 0.f;
    aq[e] = (_Float16)(qv * scale);
  }
  float mrow[8], lrow[8];
  v8f acc = {};
#pragma unroll
  for (int r = 0; r < 8; ++r) { mrow[r] = -3.0e38f; lrow[r] = 0.f; }

  __shared__ float sP[16][33];

  for (int jt = 0; jt < N / 32; ++jt) {
    int key0 = jt * 32;
    v16h bk0, bk1;
#pragma unroll
    for (int e = 0; e < 16; ++e) {
      int k = b_kmap(e, hi);
      float kv0 = (k < 8) ? (float)K[base + (size_t)(key0 + lr) * 32 + h * 8 + k] : 0.f;
      float kv1 = (k < 8) ? (float)K[base + (size_t)(key0 + 16 + lr) * 32 + h * 8 + k] : 0.f;
      bk0[e] = (_Float16)kv0;
      bk1[e] = (_Float16)kv1;
    }
    v8f z = {};
    v8f s0 = wmma_f16(aq, bk0, z);
    v8f s1 = wmma_f16(aq, bk1, z);

#pragma unroll
    for (int r = 0; r < 8; ++r) {
      float mv = fmaxf(s0[r], s1[r]);
#pragma unroll
      for (int off = 8; off; off >>= 1) mv = fmaxf(mv, __shfl_xor(mv, off));
      float mnew = fmaxf(mrow[r], mv);
      float p0 = __expf(s0[r] - mnew);
      float p1 = __expf(s1[r] - mnew);
      float ps = p0 + p1;
#pragma unroll
      for (int off = 8; off; off >>= 1) ps += __shfl_xor(ps, off);
      float fac = __expf(mrow[r] - mnew);
      lrow[r] = lrow[r] * fac + ps;
      mrow[r] = mnew;
      acc[r] *= fac;
      sP[r + 8 * hi][lr] = p0;
      sP[r + 8 * hi][16 + lr] = p1;
    }
    __syncthreads();
    v16h ap, bv;
#pragma unroll
    for (int e = 0; e < 16; ++e) {
      ap[e] = (_Float16)sP[lr][a_kmap(e, hi)];
      int kc = b_kmap(e, hi);
      float vv = (lr < 8) ? (float)V[base + (size_t)(key0 + kc) * 32 + h * 8 + lr] : 0.f;
      bv[e] = (_Float16)vv;
    }
    acc = wmma_f16(ap, bv, acc);
    __syncthreads();
    if (jt + 1 < N / 32) {
      __builtin_prefetch(&K[base + (size_t)(key0 + 32 + lr) * 32], 0, 0);
      __builtin_prefetch(&V[base + (size_t)(key0 + 32 + lr) * 32], 0, 0);
    }
  }
#pragma unroll
  for (int r = 0; r < 8; ++r) {
    if (lr < 8) {
      int rg = row0 + r + 8 * hi;
      O[base + (size_t)rg * 32 + h * 8 + lr] = acc[r] / lrow[r];
    }
  }
}

// ---------------------------------------------------------------------------
// 7) FFT kernels (naive DFT, 64-point; trivially small FLOPs).
//    xf layout: [B][64ch][64h][33kw], ch<32 real, ch>=32 imag.
// ---------------------------------------------------------------------------
__global__ void k_rfft_w(const float* __restrict__ x, float* __restrict__ xf) {
  int idx = blockIdx.x * blockDim.x + threadIdx.x;
  const int total = BB * CC * HH_ * WF;
  if (idx >= total) return;
  int kw = idx % WF; int t = idx / WF;
  int hh = t % HH_; t /= HH_;
  int c = t % CC; int b = t / CC;
  float re = 0.f, im = 0.f;
  float w0 = -TWO_PI * (float)kw / 64.f;
  for (int ww = 0; ww < 64; ++ww) {
    float v = x[((size_t)(b * HH_ + hh) * WW_ + ww) * CC + c];
    float ang = w0 * (float)ww;
    re += v * __cosf(ang);
    im += v * __sinf(ang);
  }
  xf[((size_t)(b * C2C + c) * HH_ + hh) * WF + kw] = re;
  xf[((size_t)(b * C2C + c + CC) * HH_ + hh) * WF + kw] = im;
}

__global__ void k_fft_h(const float* __restrict__ xf, float* __restrict__ xo) {
  int idx = blockIdx.x * blockDim.x + threadIdx.x;
  const int total = BB * CC * HH_ * WF;
  if (idx >= total) return;
  int kw = idx % WF; int t = idx / WF;
  int kh = t % HH_; t /= HH_;
  int c = t % CC; int b = t / CC;
  float re = 0.f, im = 0.f;
  float w0 = -TWO_PI * (float)kh / 64.f;
  for (int hh = 0; hh < 64; ++hh) {
    float a = xf[((size_t)(b * C2C + c) * HH_ + hh) * WF + kw];
    float bi = xf[((size_t)(b * C2C + c + CC) * HH_ + hh) * WF + kw];
    float ang = w0 * (float)hh, cs = __cosf(ang), sn = __sinf(ang);
    re += a * cs - bi * sn;
    im += a * sn + bi * cs;
  }
  xo[((size_t)(b * C2C + c) * HH_ + kh) * WF + kw] = re;
  xo[((size_t)(b * C2C + c + CC) * HH_ + kh) * WF + kw] = im;
}

// inverse fft along H with LeakyReLU folded into the reads; scale 1/64
__global__ void k_ifft_h(const float* __restrict__ y, float* __restrict__ yo) {
  int idx = blockIdx.x * blockDim.x + threadIdx.x;
  const int total = BB * CC * HH_ * WF;
  if (idx >= total) return;
  int kw = idx % WF; int t = idx / WF;
  int hh = t % HH_; t /= HH_;
  int c = t % CC; int b = t / CC;
  float re = 0.f, im = 0.f;
  float w0 = TWO_PI * (float)hh / 64.f;
  for (int kh = 0; kh < 64; ++kh) {
    float a = leakyf_(y[((size_t)(b * C2C + c) * HH_ + kh) * WF + kw]);
    float bi = leakyf_(y[((size_t)(b * C2C + c + CC) * HH_ + kh) * WF + kw]);
    float ang = w0 * (float)kh, cs = __cosf(ang), sn = __sinf(ang);
    re += a * cs - bi * sn;
    im += a * sn + bi * cs;
  }
  yo[((size_t)(b * C2C + c) * HH_ + hh) * WF + kw] = re * (1.f / 64.f);
  yo[((size_t)(b * C2C + c + CC) * HH_ + hh) * WF + kw] = im * (1.f / 64.f);
}

// inverse rfft along W (Hermitian), scale 1/64; output BHWC
__global__ void k_irfft_w(const float* __restrict__ y, float* __restrict__ out) {
  int idx = blockIdx.x * blockDim.x + threadIdx.x;
  const int total = BB * CC * HH_ * WW_;
  if (idx >= total) return;
  int ww = idx % WW_; int t = idx / WW_;
  int hh = t % HH_; t /= HH_;
  int c = t % CC; int b = t / CC;
  const float* R = &y[((size_t)(b * C2C + c) * HH_ + hh) * WF];
  const float* I = &y[((size_t)(b * C2C + c + CC) * HH_ + hh) * WF];
  float parity = (ww & 1) ? -1.f : 1.f;
  float acc = R[0] + parity * R[32];
  float w0 = TWO_PI * (float)ww / 64.f;
  for (int k = 1; k < 32; ++k) {
    float ang = w0 * (float)k;
    acc += 2.f * (R[k] * __cosf(ang) - I[k] * __sinf(ang));
  }
  out[((size_t)(b * HH_ + hh) * WW_ + ww) * CC + c] = acc * (1.f / 64.f);
}

// ---------------------------------------------------------------------------
// 8) 3x3 convs (BCHW in/out, pad 1).
// ---------------------------------------------------------------------------
__global__ void k_conv3x3(const float* __restrict__ in, const float* __restrict__ w,
                          const float* __restrict__ bias, float* __restrict__ out,
                          int Ci, int Co, int H, int W, int relu) {
  int idx = blockIdx.x * blockDim.x + threadIdx.x;
  int total = BB * Co * H * W;
  if (idx >= total) return;
  int ww = idx % W; int t = idx / W;
  int hh = t % H; t /= H;
  int o = t % Co; int b = t / Co;
  float acc = bias ? bias[o] : 0.f;
  for (int kh = -1; kh <= 1; ++kh) {
    int h2 = hh + kh;
    if (h2 < 0 || h2 >= H) continue;
    for (int kw = -1; kw <= 1; ++kw) {
      int w2 = ww + kw;
      if (w2 < 0 || w2 >= W) continue;
      const float* ip = in + ((size_t)(b * Ci) * H + h2) * W + w2;
      const float* wp = w + ((size_t)(o * Ci) * 3 + (kh + 1)) * 3 + (kw + 1);
      for (int i = 0; i < Ci; ++i) acc += ip[(size_t)i * H * W] * wp[i * 9];
    }
  }
  if (relu) acc = fmaxf(acc, 0.f);
  out[((size_t)(b * Co + o) * H + hh) * W + ww] = acc;
}

// decoder conv1: input = concat(xs BHWC, xfr BHWC) as 64 channels, relu
__global__ void k_dec_conv1(const float* __restrict__ xs, const float* __restrict__ xfr,
                            const float* __restrict__ w, const float* __restrict__ bias,
                            float* __restrict__ out) {
  int idx = blockIdx.x * blockDim.x + threadIdx.x;
  const int total = BB * C2C * HH_ * WW_;
  if (idx >= total) return;
  int ww = idx % WW_; int t = idx / WW_;
  int hh = t % HH_; t /= HH_;
  int o = t % C2C; int b = t / C2C;
  float acc = bias[o];
  for (int kh = -1; kh <= 1; ++kh) {
    int h2 = hh + kh;
    if (h2 < 0 || h2 >= HH_) continue;
    for (int kw = -1; kw <= 1; ++kw) {
      int w2 = ww + kw;
      if (w2 < 0 || w2 >= WW_) continue;
      size_t pbase = ((size_t)(b * HH_ + h2) * WW_ + w2) * CC;
      const float* wp = w + ((size_t)(o * C2C) * 3 + (kh + 1)) * 3 + (kw + 1);
      for (int i = 0; i < CC; ++i) acc += xs[pbase + i] * wp[i * 9];
      for (int i = 0; i < CC; ++i) acc += xfr[pbase + i] * wp[(CC + i) * 9];
    }
  }
  out[((size_t)(b * C2C + o) * HH_ + hh) * WW_ + ww] = fmaxf(acc, 0.f);
}

// ---------------------------------------------------------------------------
// Host side
// ---------------------------------------------------------------------------
struct KanP {
  const float *bw, *bb, *b2w, *b2b, *sw, *sb, *s2w, *s2b;
  const float *k1b, *k1s, *k2b, *k2s, *q1b, *q1s, *q2b, *q2s;
};

static KanP parse_kb(void* const* d, int& i) {
  KanP k;
  k.bw  = (const float*)d[i++]; k.bb  = (const float*)d[i++];
  k.b2w = (const float*)d[i++]; k.b2b = (const float*)d[i++];
  k.sw  = (const float*)d[i++]; k.sb  = (const float*)d[i++];
  k.s2w = (const float*)d[i++]; k.s2b = (const float*)d[i++];
  k.k1b = (const float*)d[i++]; k.k1s = (const float*)d[i++];
  k.k2b = (const float*)d[i++]; k.k2s = (const float*)d[i++];
  k.q1b = (const float*)d[i++]; k.q1s = (const float*)d[i++];
  k.q2b = (const float*)d[i++]; k.q2s = (const float*)d[i++];
  return k;
}

extern "C" void kernel_launch(void* const* d_in, const int* in_sizes, int n_in,
                              void* d_out, int out_size, void* d_ws, size_t ws_size,
                              hipStream_t stream) {
  (void)in_sizes; (void)n_in; (void)out_size; (void)ws_size;
  // ---- parse leaves (setup_inputs insertion order, recursive) -------------
  int i = 0;
  const float* input_x  = (const float*)d_in[i++];
  const float* input_b  = (const float*)d_in[i++];
  const float* simple_w = (const float*)d_in[i++];
  const float* ln_g     = (const float*)d_in[i++];
  const float* ln_b     = (const float*)d_in[i++];
  const float* W_q      = (const float*)d_in[i++];
  const float* W_k      = (const float*)d_in[i++];
  const float* W_v      = (const float*)d_in[i++];
  const float* W_o      = (const float*)d_in[i++];
  KanP kb2_0 = parse_kb(d_in, i);
  KanP kb2_1 = parse_kb(d_in, i);
  KanP kb2_2 = parse_kb(d_in, i); (void)kb2_2;  // declared but unused in fwd
  KanP kbf0 = parse_kb(d_in, i);
  KanP kbf1 = parse_kb(d_in, i);
  KanP kbf2 = parse_kb(d_in, i);
  const float* freq_w = (const float*)d_in[i++];
  const float* freq_b = (const float*)d_in[i++];
  const float* dec_w1 = (const float*)d_in[i++];
  const float* dec_b1 = (const float*)d_in[i++];
  const float* dec_w2 = (const float*)d_in[i++];
  const float* dec_b2 = (const float*)d_in[i++];

  // ---- workspace bump allocator -------------------------------------------
  char* ws = (char*)d_ws;
  size_t off = 0;
  auto alloc = [&](size_t bytes) -> void* {
    void* p = ws + off;
    off = (off + bytes + 255) & ~(size_t)255;
    return p;
  };
  const size_t NF  = (size_t)NROW * CC;          // 262144 activations
  float* x_base   = (float*)alloc(NF * 4);
  float* bufS1    = (float*)alloc(NF * 4);
  float* bufS2    = (float*)alloc(NF * 4);
  float* bufF1    = (float*)alloc(NF * 4);
  float* bufF2    = (float*)alloc(NF * 4);
  float* t_ln     = (float*)alloc(NF * 4);
  float* f_buf    = (float*)alloc(NF * 4);
  float* fq       = (float*)alloc(NF * 4);
  float* attn_out = (float*)alloc(NF * 4);
  _Float16* q16   = (_Float16*)alloc(NF * 2);
  _Float16* k16   = (_Float16*)alloc(NF * 2);
  _Float16* v16   = (_Float16*)alloc(NF * 2);
  const size_t NFQ = (size_t)BB * C2C * PFREQ;   // 270336 freq activations
  float* xf    = (float*)alloc(NFQ * 4);
  float* xf2   = (float*)alloc(NFQ * 4);
  float* yfreq = (float*)alloc(NFQ * 4);
  float* yifft = (float*)alloc(NFQ * 4);
  float* y1    = (float*)alloc((size_t)BB * C2C * NPIX * 4);
  float* pooled = (float*)alloc(BB * C2C * 4);
  float* csb    = (float*)alloc(BB * C2C * 4);
  float* ssb    = (float*)alloc(BB * C2C * 4);

  auto kan_block = [&](const float* x, int C, int P, int bhwc, const KanP& kb,
                       const float* res, float* out) {
    int R = (C == 32) ? 4 : 8;
    k_pool<<<dim3(2 * C), dim3(256), 0, stream>>>(x, pooled, C, P, bhwc);
    k_kan_gates<<<dim3(1), dim3(2 * C), 0, stream>>>(
        pooled, kb.bw, kb.bb, kb.b2w, kb.b2b, kb.k1b, kb.k1s, kb.k2b, kb.k2s,
        csb, C, R);
    k_kan_gates<<<dim3(1), dim3(2 * C), 0, stream>>>(
        pooled, kb.sw, kb.sb, kb.s2w, kb.s2b, kb.q1b, kb.q1s, kb.q2b, kb.q2s,
        ssb, C, R);
    int total = BB * C * P;
    k_kan_apply<<<dim3((total + 255) / 256), dim3(256), 0, stream>>>(
        x, csb, ssb, res, out, C, P, bhwc);
  };

  // ---- stem ---------------------------------------------------------------
  k_conv1x1<<<dim3(NROW / 16), dim3(32), 0, stream>>>(input_x, input_b, simple_w, x_base);

  // ---- spatial branch x3 --------------------------------------------------
  {
    const float* cur = x_base;
    float* nxt = bufS1;
    for (int it = 0; it < 3; ++it) {
      k_layernorm32<<<dim3(NROW / 8), dim3(256), 0, stream>>>(cur, ln_g, ln_b, t_ln);
      kan_block(t_ln, CC, NPIX, 1, kb2_0, nullptr, f_buf);
      k_proj32_f16<<<dim3(NROW / 16), dim3(32), 0, stream>>>(f_buf, W_q, q16);
      k_proj32_f16<<<dim3(NROW / 16), dim3(32), 0, stream>>>(f_buf, W_k, k16);
      k_proj32_f16<<<dim3(NROW / 16), dim3(32), 0, stream>>>(f_buf, W_v, v16);
      k_attn_flash<<<dim3(NPIX / 16, 4, BB), dim3(32), 0, stream>>>(q16, k16, v16, attn_out);
      k_outproj32<<<dim3(NROW / 16), dim3(32), 0, stream>>>(attn_out, W_o, f_buf, fq);
      k_layernorm32<<<dim3(NROW / 8), dim3(256), 0, stream>>>(fq, ln_g, ln_b, t_ln);
      kan_block(t_ln, CC, NPIX, 1, kb2_1, fq, nxt);
      cur = nxt;
      nxt = (cur == bufS1) ? bufS2 : bufS1;
    }
  }
  const float* xs_final = bufS1;  // after 3 ping-pongs: base->S1->S2->S1

  // ---- frequency branch x3 ------------------------------------------------
  {
    const float* cur = x_base;
    float* nxt = bufF1;
    const KanP* kbl[3] = {&kbf0, &kbf1, &kbf2};
    const int NT = BB * CC * HH_ * WF;
    for (int it = 0; it < 3; ++it) {
      k_rfft_w<<<dim3((NT + 255) / 256), dim3(256), 0, stream>>>(cur, xf);
      k_fft_h<<<dim3((NT + 255) / 256), dim3(256), 0, stream>>>(xf, xf2);
      {
        int total = BB * C2C * HH_ * WF;
        k_conv3x3<<<dim3((total + 255) / 256), dim3(256), 0, stream>>>(
            xf2, freq_w, freq_b, yfreq, C2C, C2C, HH_, WF, 0);
      }
      for (int t = 0; t < 3; ++t)
        kan_block(yfreq, C2C, PFREQ, 0, *kbl[t], nullptr, yfreq);
      k_ifft_h<<<dim3((NT + 255) / 256), dim3(256), 0, stream>>>(yfreq, yifft);
      {
        int total = BB * CC * HH_ * WW_;
        k_irfft_w<<<dim3((total + 255) / 256), dim3(256), 0, stream>>>(yifft, nxt);
      }
      cur = nxt;
      nxt = (cur == bufF1) ? bufF2 : bufF1;
    }
  }
  const float* xfr_final = bufF1;

  // ---- decoder ------------------------------------------------------------
  {
    int total = BB * C2C * HH_ * WW_;
    k_dec_conv1<<<dim3((total + 255) / 256), dim3(256), 0, stream>>>(
        xs_final, xfr_final, dec_w1, dec_b1, y1);
    k_conv3x3<<<dim3((total + 255) / 256), dim3(256), 0, stream>>>(
        y1, dec_w2, dec_b2, (float*)d_out, C2C, C2C, HH_, WW_, 0);
  }
}
